// Encoder_58377195487822
// MI455X (gfx1250) — 16-bit and 8-bit matrix operand order repaired and checked by reading, compiled, not run on hardware
//
#include <hip/hip_runtime.h>
#include <hip/hip_bf16.h>


typedef __attribute__((ext_vector_type(16))) _Float16 v16h;
typedef __attribute__((ext_vector_type(8)))  _Float16 v8h;
typedef __attribute__((ext_vector_type(8)))  float    v8f;
typedef __attribute__((ext_vector_type(8)))  int      v8i;

#define DEVINL __device__ __attribute__((always_inline))

constexpr int Bq = 512, Tq = 256, Dq = 81, Hq = 128;
constexpr int NTHREADS = 256;
constexpr int FRAG_H = 512;
constexpr int FRAG_B = 1024;

constexpr int OFF_W1  = 0;                    constexpr int SZ_W1  = 8  * 8  * FRAG_B;
constexpr int OFF_W2  = OFF_W1  + SZ_W1;      constexpr int SZ_W2  = 4  * 6  * FRAG_B;
constexpr int OFF_WIH = OFF_W2  + SZ_W2;      constexpr int SZ_WIH = 3  * 32 * FRAG_B;
constexpr int OFF_WHH = OFF_WIH + SZ_WIH;     constexpr int SZ_WHH = 2  * 32 * FRAG_B;
constexpr int OFF_HS  = OFF_WHH + SZ_WHH;     constexpr int SZ_HS  = 16 * 256 * 2;
constexpr int OFF_HC8 = OFF_HS  + SZ_HS;      constexpr int SZ_HC8 = 16 * 128;
constexpr int OFF_Z   = OFF_HC8 + SZ_HC8;     constexpr int SZ_Z   = 16 * 128 * 2;
constexpr int OFF_E   = OFF_Z   + SZ_Z;       constexpr int SZ_E   = 16 * 96  * 4;
constexpr int OFF_XW  = OFF_E   + SZ_E;       constexpr int SZ_XW  = 16 * 96  * 2;
constexpr int OFF_RED = OFF_XW  + SZ_XW;      constexpr int SZ_RED = 16 * 16  * 4;
constexpr int SMEM_BYTES = OFF_RED + SZ_RED;

constexpr float LOG2E = 1.44269504088896340736f;

DEVINL float fast_exp(float x) {
  return __builtin_amdgcn_exp2f(LOG2E * x);
}
DEVINL float fast_sigm(float x) {
  return __builtin_amdgcn_rcpf(1.0f + __builtin_amdgcn_exp2f(-LOG2E * x));
}
DEVINL float fast_tanh(float x) {
  float e = __builtin_amdgcn_exp2f(-2.0f * LOG2E * __builtin_fabsf(x));
  float r = (1.0f - e) * __builtin_amdgcn_rcpf(1.0f + e);
  return __builtin_copysignf(r, x);
}


DEVINL v16h loadA16(const _Float16* p, int lda, int k0) {
  int lane = threadIdx.x & 31;
  const _Float16* row = p + (lane & 15) * lda + k0 + ((lane >> 4) << 3);
  v8h lo = *(const v8h*)(row);
  v8h hi = *(const v8h*)(row + 16);
  return __builtin_shufflevector(lo, hi, 0,1,2,3,4,5,6,7,8,9,10,11,12,13,14,15);
}

DEVINL v8i loadA8(const unsigned char* p, int lda, int k0) {
  int lane = threadIdx.x & 31;
  const unsigned char* row = p + (lane & 15) * lda + k0 + ((lane >> 4) << 3);
  v8i a;
#pragma unroll
  for (int g = 0; g < 4; ++g) {
    uint2 d = *(const uint2*)(row + g * 16);
    a[2 * g] = (int)d.x; a[2 * g + 1] = (int)d.y;
  }
  return a;
}

DEVINL v16h loadB16(const _Float16* frag) {
  int lane = threadIdx.x & 31;
  const v8h* p = (const v8h*)(frag + lane * 16);
  return __builtin_shufflevector(p[0], p[1], 0,1,2,3,4,5,6,7,8,9,10,11,12,13,14,15);
}
DEVINL v8i loadB8(const unsigned char* frag) {
  int lane = threadIdx.x & 31;
  return *(const v8i*)(frag + lane * 32);
}

DEVINL unsigned char f32_to_e4m3(float x) {
  unsigned u = __float_as_uint(x);
  unsigned s = (u >> 31) << 7;
  unsigned be = (u >> 23) & 0xFF;
  if (be == 0xFF) return (unsigned char)(s | 0x7E);
  int E = (int)be - 127 + 7;
  if (E >= 16) return (unsigned char)(s | 0x7E);
  if (E >= 1) {
    unsigned man = u & 0x7FFFFF;
    unsigned m = man >> 20, rem = man & 0xFFFFF;
    if (rem > 0x80000u || (rem == 0x80000u && (m & 1))) ++m;
    if (m == 8) { m = 0; if (++E >= 16) return (unsigned char)(s | 0x7E); }
    return (unsigned char)(s | ((unsigned)E << 3) | m);
  }
  float ax = __uint_as_float(u & 0x7FFFFFFFu);
  int m = (int)(ax * 512.0f + 0.5f);
  if (m >= 8) return (unsigned char)(s | (1u << 3));
  return (unsigned char)(s | (unsigned)m);
}

DEVINL void stage_b16(_Float16* dst, const float* src, int ld, int Ksrc, int Nsrc,
                      int KC, int NT, int tid) {
  int total = KC * NT * FRAG_H;
  for (int idx = tid; idx < total; idx += NTHREADS) {
    int frag = idx >> 9, r = idx & 511;
    int lane = r >> 4, j = r & 15;
    int kc = frag / NT, nt = frag - kc * NT;
    int k = kc * 32 + ((lane >> 4) << 3) + ((j >> 3) << 4) + (j & 7);
    int n = nt * 16 + (lane & 15);
    float w = (k < Ksrc && n < Nsrc) ? src[n * ld + k] : 0.0f;
    dst[idx] = (_Float16)w;
  }
}

__global__ __launch_bounds__(NTHREADS, 1)
void Encoder_58377195487822_kernel(const float* __restrict__ X,
                                   const float* __restrict__ W1,  const float* __restrict__ b1,
                                   const float* __restrict__ W2,  const float* __restrict__ b2,
                                   const float* __restrict__ Wih, const float* __restrict__ Whh,
                                   const float* __restrict__ bih, const float* __restrict__ bhh,
                                   float* __restrict__ h_out, float* __restrict__ a_out) {
  extern __shared__ char smem[];
  _Float16*      w1f  = (_Float16*)     (smem + OFF_W1);
  _Float16*      w2f  = (_Float16*)     (smem + OFF_W2);
  _Float16*      wihf = (_Float16*)     (smem + OFF_WIH);
  unsigned char* whhf = (unsigned char*)(smem + OFF_WHH);
  _Float16*      hsb  = (_Float16*)     (smem + OFF_HS);
  unsigned char* hc8  = (unsigned char*)(smem + OFF_HC8);
  _Float16*      zb   = (_Float16*)     (smem + OFF_Z);
  float*         eb   = (float*)        (smem + OFF_E);
  _Float16*      xwb  = (_Float16*)     (smem + OFF_XW);
  float*         red  = (float*)        (smem + OFF_RED);

  const int tid  = threadIdx.x;
  const int wv   = tid >> 5;
  const int lane = tid & 31;
  const int cl   = lane & 15;
  const int mh   = (lane >> 4) << 3;
  const int b0   = blockIdx.x * 16;

  stage_b16(w1f,  W1,  2 * Hq, 2 * Hq, Hq,  8,  8, tid);
  stage_b16(w2f,  W2,  Hq,     Hq,     Dq,  4,  6, tid);
  stage_b16(wihf, Wih, Dq,     Dq,     4*Hq,3, 32, tid);
  {
    for (int idx = tid; idx < 2 * 32 * FRAG_B; idx += NTHREADS) {
      int frag = idx >> 10, q0 = idx & 1023;
      int ln = q0 >> 5, q = q0 & 31;
      int kc = frag >> 5, nt = frag & 31;
      int k = kc * 64 + ((q >> 3) << 4) + ((ln >> 4) << 3) + (q & 7);
      int n = nt * 16 + (ln & 15);
      whhf[idx] = f32_to_e4m3(Whh[n * Hq + k]);
    }
  }
  for (int i = tid; i < 16 * 256; i += NTHREADS) hsb[i] = (_Float16)0.0f;
  for (int i = tid; i < 16 * 128; i += NTHREADS) hc8[i] = 0;

  const int nnA = wv * 16 + cl;
  const float bias1 = b1[nnA];
  const float bias2 = (wv < 6 && nnA < Dq) ? b2[nnA] : 0.0f;
  float biasg[4];
#pragma unroll
  for (int g = 0; g < 4; ++g) biasg[g] = bih[g * Hq + nnA] + bhh[g * Hq + nnA];

  v8f creg;
#pragma unroll
  for (int r = 0; r < 8; ++r) creg[r] = 0.0f;
  __syncthreads();

  for (int t = 0; t < Tq; ++t) {
    {
      v8f acc;
#pragma unroll
      for (int r = 0; r < 8; ++r) acc[r] = bias1;
#pragma unroll
      for (int kc = 0; kc < 8; ++kc) {
        v16h a = loadA16(hsb, 256, kc * 32);
        v16h b = loadB16(w1f + (kc * 8 + wv) * FRAG_H);
        acc = __builtin_amdgcn_wmma_f32_16x16x32_f16(false, a, false, b,
                                                     (short)0, acc, false, false);
      }
#pragma unroll
      for (int r = 0; r < 8; ++r) zb[(mh + r) * 128 + nnA] = (_Float16)fast_tanh(acc[r]);
    }
    __syncthreads();

    if (wv < 6) {
      v8f acc;
#pragma unroll
      for (int r = 0; r < 8; ++r) acc[r] = bias2;
#pragma unroll
      for (int kc = 0; kc < 4; ++kc) {
        v16h a = loadA16(zb, 128, kc * 32);
        v16h b = loadB16(w2f + (kc * 6 + wv) * FRAG_H);
        acc = __builtin_amdgcn_wmma_f32_16x16x32_f16(false, a, false, b,
                                                     (short)0, acc, false, false);
      }
#pragma unroll
      for (int r = 0; r < 8; ++r) eb[(mh + r) * 96 + nnA] = acc[r];
    } else if (t + 1 < Tq) {
      int q = tid - 192;
      int row = q & 15, seg = q >> 4;
      __builtin_prefetch(X + ((size_t)(b0 + row) * Tq + (t + 1)) * Dq + seg * 21, 0, 1);
    }
    __syncthreads();

    {
      int row = tid >> 4, ct = tid & 15;
      const float* xrow = X + ((size_t)(b0 + row) * Tq + t) * Dq;
      float vals[6]; float mx = -1e30f;
#pragma unroll
      for (int jj = 0; jj < 6; ++jj) {
        int col = ct + 16 * jj;
        float v = (col < Dq) ? eb[row * 96 + col] : -1e30f;
        vals[jj] = v; mx = fmaxf(mx, v);
      }
      red[row * 16 + ct] = mx;
      __syncthreads();
      float rmx = -1e30f;
#pragma unroll
      for (int q = 0; q < 16; ++q) rmx = fmaxf(rmx, red[row * 16 + q]);
      __syncthreads();
      float s = 0.0f;
#pragma unroll
      for (int jj = 0; jj < 6; ++jj) {
        int col = ct + 16 * jj;
        if (col < Dq) { float p = fast_exp(vals[jj] - rmx); vals[jj] = p; s += p; }
      }
      red[row * 16 + ct] = s;
      __syncthreads();
      float tot = 0.0f;
#pragma unroll
      for (int q = 0; q < 16; ++q) tot += red[row * 16 + q];
      float inv = __builtin_amdgcn_rcpf(tot);
#pragma unroll
      for (int jj = 0; jj < 6; ++jj) {
        int col = ct + 16 * jj;
        if (col < Dq) {
          float al = vals[jj] * inv;
          a_out[((size_t)(b0 + row) * Tq + t) * Dq + col] = al;
          xwb[row * 96 + col] = (_Float16)(al * xrow[col]);
        } else {
          xwb[row * 96 + col] = (_Float16)0.0f;
        }
      }
    }
    __syncthreads();

    {
      v8f acc[4];
#pragma unroll
      for (int g = 0; g < 4; ++g) {
#pragma unroll
        for (int r = 0; r < 8; ++r) acc[g][r] = biasg[g];
      }
#pragma unroll
      for (int kc = 0; kc < 3; ++kc) {
        v16h a = loadA16(xwb, 96, kc * 32);
#pragma unroll
        for (int g = 0; g < 4; ++g) {
          v16h b = loadB16(wihf + (kc * 32 + g * 8 + wv) * FRAG_H);
          acc[g] = __builtin_amdgcn_wmma_f32_16x16x32_f16(false, a, false, b,
                                                          (short)0, acc[g], false, false);
        }
      }
#pragma unroll
      for (int kc = 0; kc < 2; ++kc) {
        v8i a = loadA8(hc8, 128, kc * 64);
#pragma unroll
        for (int g = 0; g < 4; ++g) {
          v8i b = loadB8(whhf + (kc * 32 + g * 8 + wv) * FRAG_B);
          acc[g] = __builtin_amdgcn_wmma_f32_16x16x64_fp8_fp8(a, b, (short)0,
                                                              acc[g], false, false);
        }
      }
#pragma unroll
      for (int r = 0; r < 8; ++r) {
        int m = mh + r;
        float ig = fast_sigm(acc[0][r]);
        float fg = fast_sigm(acc[1][r]);
        float gg = fast_tanh(acc[2][r]);
        float og = fast_sigm(acc[3][r]);
        float cn = fg * creg[r] + ig * gg;
        creg[r] = cn;
        float hn = og * fast_tanh(cn);
        hsb[m * 256 + nnA]       = (_Float16)hn;
        hsb[m * 256 + 128 + nnA] = (_Float16)cn;
        hc8[m * 128 + nnA]       = f32_to_e4m3(hn);
        h_out[((size_t)(b0 + m) * Tq + t) * Hq + nnA] = hn;
      }
    }
    __syncthreads();
  }
}

extern "C" void kernel_launch(void* const* d_in, const int* in_sizes, int n_in,
                              void* d_out, int out_size, void* d_ws, size_t ws_size,
                              hipStream_t stream) {
  const float* X   = (const float*)d_in[0];
  const float* W1  = (const float*)d_in[1];
  const float* b1  = (const float*)d_in[2];
  const float* W2  = (const float*)d_in[3];
  const float* b2  = (const float*)d_in[4];
  const float* Wih = (const float*)d_in[5];
  const float* Whh = (const float*)d_in[6];
  const float* bih = (const float*)d_in[7];
  const float* bhh = (const float*)d_in[8];
  float* h_out = (float*)d_out;
  float* a_out = h_out + (size_t)Bq * Tq * Hq;

  (void)hipFuncSetAttribute((const void*)Encoder_58377195487822_kernel,
                            hipFuncAttributeMaxDynamicSharedMemorySize, SMEM_BYTES);
  Encoder_58377195487822_kernel<<<dim3(Bq / 16), dim3(NTHREADS), SMEM_BYTES, stream>>>(
      X, W1, b1, W2, b2, Wih, Whh, bih, bhh, h_out, a_out);
}
